// FMRM_4655744548973
// MI455X (gfx1250) — matrix operand order repaired and checked by reading, not compiled, not run on hardware
//
#include <hip/hip_runtime.h>
#include <hip/hip_bf16.h>

// ---------------------------------------------------------------------------
// Problem constants
// ---------------------------------------------------------------------------
#define HIDDEN 640
#define INNER  64
#define P      25      // patches (5x5)
#define PP     32      // padded patches for WMMA tiles
#define NA     64      // features_a batch
#define NB     2048    // features_b batch
#define FEAT_ELEMS (HIDDEN * P)        // 16000 floats per batch element
#define QKV_STRIDE (PP * INNER)        // 2048 elems per (batch) q/k/v tensor

typedef __attribute__((ext_vector_type(16))) __bf16 v16bf;
typedef __attribute__((ext_vector_type(8)))  float  v8f;

// ---------------------------------------------------------------------------
// WMMA helpers (gfx1250 layouts, wave32)
// ---------------------------------------------------------------------------
__device__ __forceinline__ v8f wmma_bf16(v16bf a, v16bf b, v8f c) {
    // D = A(16x32 bf16) * B(32x16 bf16) + C(16x16 f32)
    return __builtin_amdgcn_wmma_f32_16x16x32_bf16(
        false, a, false, b, (short)0, c, false, false);
}

// A-fragment: rows [m0, m0+16), K-cols [k0, k0+32) of row-major bf16 [ld]
// Layout: lanes 0-15 row m, K0..15;
//         lanes 16-31 same rows, K16..31.
template <typename PT>
__device__ __forceinline__ v16bf load_a(PT src, int ld, int m0, int k0, int lane) {
    const int half = lane >> 4;
    const int m    = m0 + (lane & 15);
    PT row = src + m * ld + k0;
    v16bf a;
#pragma unroll
    for (int e = 0; e < 16; e += 2) {
        const int k = half * 16 + e;
        a[e]     = row[k];
        a[e + 1] = row[k + 1];
    }
    return a;
}

// B-fragment from a TRANSPOSED (column-major) store: element(k,n) =
// src[(n0+n)*ld + k0+k]  (contiguous in k -> vectorizes to b128).
// Lanes 0-15: K0..15, lanes 16-31: K16..31.
template <typename PT>
__device__ __forceinline__ v16bf load_bt(PT src, int ld, int k0, int n0, int lane) {
    const int n  = n0 + (lane & 15);
    const int kc = k0 + ((lane >> 4) << 4);
    PT p = src + n * ld + kc;
    v16bf b;
#pragma unroll
    for (int e = 0; e < 16; ++e) b[e] = p[e];
    return b;
}

// D-tile store (16x16 f32): VGPR i -> row m0+8*half+i, col n0+(lane&15)
template <typename PT>
__device__ __forceinline__ void store_d(PT dst, int ld, int m0, int n0, int lane, v8f d) {
    const int half = lane >> 4;
    const int n    = n0 + (lane & 15);
#pragma unroll
    for (int i = 0; i < 8; ++i) dst[(m0 + half * 8 + i) * ld + n] = d[i];
}

__device__ __forceinline__ v8f zero8() {
    v8f z = {0.f,0.f,0.f,0.f,0.f,0.f,0.f,0.f};
    return z;
}

// ---------------------------------------------------------------------------
// One attention direction for one (q,w) pair, executed by a single wave.
//   S   = Q(25x64) @ K(25x64)^T      (padded 32x32, bf16 WMMA, f32 acc)
//   Pm  = softmax(S * 0.125) masked to 25 cols
//   rec = Pm @ V(25x64)              (padded 32x64; V given TRANSPOSED 64x32)
//   ret = sum_{p<25,i} (Vref[p,i] - rec[p,i])^2   (reduced across the wave)
// Template over pointer types so LDS (__shared__) vs global address spaces
// survive inlining -> ds_* vs global_* instructions.
// ---------------------------------------------------------------------------
template <typename PQ, typename PK, typename PV, typename PVR>
__device__ __forceinline__ float attn_dir(PQ Q, PK K, PV VT, PVR Vref,
                                          float* sS, __bf16* sP, int lane) {
    // ---- S = Q @ K^T ------------------------------------------------------
    v8f acc[2][2];
#pragma unroll
    for (int i = 0; i < 2; ++i)
#pragma unroll
        for (int j = 0; j < 2; ++j) acc[i][j] = zero8();

#pragma unroll
    for (int kc = 0; kc < INNER; kc += 32) {
        v16bf a0 = load_a (Q, INNER,  0, kc, lane);
        v16bf a1 = load_a (Q, INNER, 16, kc, lane);
        v16bf b0 = load_bt(K, INNER, kc,  0, lane);   // B(k,n) = K[n][k]
        v16bf b1 = load_bt(K, INNER, kc, 16, lane);
        acc[0][0] = wmma_bf16(a0, b0, acc[0][0]);
        acc[0][1] = wmma_bf16(a0, b1, acc[0][1]);
        acc[1][0] = wmma_bf16(a1, b0, acc[1][0]);
        acc[1][1] = wmma_bf16(a1, b1, acc[1][1]);
    }
    store_d(sS, PP,  0,  0, lane, acc[0][0]);
    store_d(sS, PP,  0, 16, lane, acc[0][1]);
    store_d(sS, PP, 16,  0, lane, acc[1][0]);
    store_d(sS, PP, 16, 16, lane, acc[1][1]);
    __syncthreads();

    // ---- row softmax (lane r handles row r; cols >= 25 masked out) --------
    {
        const float* row = sS + lane * PP;
        float mx = -1e30f;
#pragma unroll
        for (int k = 0; k < P; ++k) mx = fmaxf(mx, row[k]);
        float e[P];
        float sum = 0.f;
#pragma unroll
        for (int k = 0; k < P; ++k) { e[k] = __expf((row[k] - mx) * 0.125f); sum += e[k]; }
        const float inv = 1.f / sum;
        __bf16* prow = sP + lane * PP;
#pragma unroll
        for (int k = 0; k < P; ++k) prow[k] = (__bf16)(e[k] * inv);
#pragma unroll
        for (int k = P; k < PP; ++k) prow[k] = (__bf16)0.f;
    }
    __syncthreads();

    // ---- rec = Pm @ V  (32x64); V supplied transposed [64][32] ------------
    v16bf pa0 = load_a(sP, PP,  0, 0, lane);
    v16bf pa1 = load_a(sP, PP, 16, 0, lane);
    v8f r[2][4];
#pragma unroll
    for (int ct = 0; ct < 4; ++ct) {
        v16bf bv = load_bt(VT, PP, 0, ct * 16, lane);  // B(k,n) = VT[n][k] = V[k][n]
        r[0][ct] = wmma_bf16(pa0, bv, zero8());
        r[1][ct] = wmma_bf16(pa1, bv, zero8());
    }

    // ---- squared distance vs Vref (rows < 25 only) ------------------------
    const int half = lane >> 4, nl = lane & 15;
    float part = 0.f;
#pragma unroll
    for (int rt = 0; rt < 2; ++rt)
#pragma unroll
        for (int ct = 0; ct < 4; ++ct)
#pragma unroll
            for (int i = 0; i < 8; ++i) {
                const int m = rt * 16 + half * 8 + i;
                if (m < P) {
                    const float d = Vref[m * INNER + ct * 16 + nl] - r[rt][ct][i];
                    part += d * d;
                }
            }
#pragma unroll
    for (int off = 16; off > 0; off >>= 1) part += __shfl_xor(part, off, 32);
    return part;
}

// ---------------------------------------------------------------------------
// Phase 1: QKV projection. One block per batch element (2112 blocks).
// x[p,c] = feat[c,p];  qkv[p,:] = x[p,:] @ W (640x192); split into q,k,v.
// q,k stored row-major [PP][INNER] bf16; v stored TRANSPOSED [INNER][PP] bf16
// (so P@V B-fragments read contiguously) plus a row-major f32 copy for the
// distance term. All pads zeroed.
// ---------------------------------------------------------------------------
__global__ __launch_bounds__(256) void qkv_kernel(
    const float* __restrict__ fa, const float* __restrict__ fb,
    const float* __restrict__ W,
    __bf16* qa, __bf16* ka, __bf16* vaT, float* vaf,
    __bf16* qb, __bf16* kb, __bf16* vbT, float* vbf) {
    __shared__ float sx[FEAT_ELEMS];   // 640*25 floats = 64000 B
    const int b   = blockIdx.x;
    const int tid = threadIdx.x;
    const bool isA = (b < NA);
    const float* feat = isA ? (fa + (size_t)b * FEAT_ELEMS)
                            : (fb + (size_t)(b - NA) * FEAT_ELEMS);
    for (int i = tid; i < FEAT_ELEMS; i += 256) sx[i] = feat[i];
    __syncthreads();

    const int idx = isA ? b : (b - NA);
    __bf16* Q  = (isA ? qa  : qb ) + (size_t)idx * QKV_STRIDE;
    __bf16* K  = (isA ? ka  : kb ) + (size_t)idx * QKV_STRIDE;
    __bf16* VT = (isA ? vaT : vbT) + (size_t)idx * QKV_STRIDE;
    float*  VF = (isA ? vaf : vbf) + (size_t)idx * QKV_STRIDE;

    for (int o = tid; o < P * 3 * INNER; o += 256) {
        const int p = o / (3 * INNER);
        const int j = o - p * (3 * INNER);
        const float* wp = W + j;
        const float* xp = sx + p;
        float s = 0.f;
#pragma unroll 4
        for (int c = 0; c < HIDDEN; ++c) s += xp[c * P] * wp[c * 3 * INNER];
        if (j < INNER)            Q[p * INNER + j] = (__bf16)s;
        else if (j < 2 * INNER)   K[p * INNER + (j - INNER)] = (__bf16)s;
        else { const int ch = j - 2 * INNER;
               VT[ch * PP + p]      = (__bf16)s;    // transposed
               VF[p * INNER + ch]   = s; }
    }
    // zero the pads: Q/K rows 25..31, VT cols 25..31, VF rows 25..31
    for (int o = tid; o < (PP - P) * INNER; o += 256) {
        const int r = P + o / INNER, c = o % INNER;
        Q[r * INNER + c]  = (__bf16)0.f;
        K[r * INNER + c]  = (__bf16)0.f;
        VT[c * PP + r]    = (__bf16)0.f;
        VF[r * INNER + c] = 0.f;
    }
}

// ---------------------------------------------------------------------------
// Phase 2: all-pairs attention + negative squared L2.
// grid = (64 windows, 64 query strips), block = 128 threads (4 waves).
// Each block stages one window's q/k/vT (+ v f32) in LDS and iterates 8x over
// 4 queries (one per wave).
// ---------------------------------------------------------------------------
__global__ __launch_bounds__(128) void attn_kernel(
    const __bf16* __restrict__ qa, const __bf16* __restrict__ ka,
    const __bf16* __restrict__ vaT, const float* __restrict__ vaf,
    const __bf16* __restrict__ qb, const __bf16* __restrict__ kb,
    const __bf16* __restrict__ vbT, const float* __restrict__ vbf,
    float* __restrict__ out) {
    __shared__ __bf16 s_qa[QKV_STRIDE];
    __shared__ __bf16 s_ka[QKV_STRIDE];
    __shared__ __bf16 s_vaT[QKV_STRIDE];
    __shared__ float  s_vaf[QKV_STRIDE];
    __shared__ float  sS[4][PP * PP];
    __shared__ __bf16 sP[4][PP * PP];

    const int w    = blockIdx.x;
    const int tid  = threadIdx.x;
    const int lane = tid & 31;
    const int wv   = tid >> 5;

    for (int i = tid; i < QKV_STRIDE; i += 128) {
        s_qa[i]  = qa [(size_t)w * QKV_STRIDE + i];
        s_ka[i]  = ka [(size_t)w * QKV_STRIDE + i];
        s_vaT[i] = vaT[(size_t)w * QKV_STRIDE + i];
        s_vaf[i] = vaf[(size_t)w * QKV_STRIDE + i];
    }
    __syncthreads();

    for (int it = 0; it < 8; ++it) {
        const int q = blockIdx.y * 32 + it * 4 + wv;
        if (it + 1 < 8) {
            // gfx1250 global_prefetch_b8, WGP scope (locality=3): pull the
            // next query's tensors into all cache levels ahead of use.
            const size_t qn = (size_t)(q + 4) * QKV_STRIDE;
            __builtin_prefetch(qb  + qn, 0, 3);
            __builtin_prefetch(kb  + qn, 0, 3);
            __builtin_prefetch(vbT + qn, 0, 3);
            __builtin_prefetch(vbf + qn, 0, 3);
        }
        // rec_b = softmax(q_b k_a^T) v_a ; compare vs v_b  -> qs_similarity
        const float qs = attn_dir(qb + (size_t)q * QKV_STRIDE,
                                  (const __bf16*)s_ka, (const __bf16*)s_vaT,
                                  vbf + (size_t)q * QKV_STRIDE,
                                  sS[wv], sP[wv], lane);
        // rec_a = softmax(q_a k_b^T) v_b ; compare vs v_a  -> sq_similarity
        const float sq = attn_dir((const __bf16*)s_qa,
                                  kb + (size_t)q * QKV_STRIDE,
                                  vbT + (size_t)q * QKV_STRIDE,
                                  (const float*)s_vaf,
                                  sS[wv], sP[wv], lane);
        if (lane == 0) {
            out[(size_t)q * NA + w]            = -sq;   // sq_similarity [Nq,Nw]
            out[(size_t)NB * NA + q * NA + w]  = -qs;   // qs_similarity [Nq,Nw]
        }
    }
}

// ---------------------------------------------------------------------------
// Host launcher
// ---------------------------------------------------------------------------
extern "C" void kernel_launch(void* const* d_in, const int* in_sizes, int n_in,
                              void* d_out, int out_size, void* d_ws, size_t ws_size,
                              hipStream_t stream) {
    const float* fa = (const float*)d_in[0];   // [64, 640, 5, 5]
    const float* fb = (const float*)d_in[1];   // [2048, 640, 5, 5]
    const float* W  = (const float*)d_in[2];   // [640, 192]

    char* ws = (char*)d_ws;
    size_t off = 0;
    __bf16* qa  = (__bf16*)(ws + off); off += (size_t)NA * QKV_STRIDE * 2;
    __bf16* ka  = (__bf16*)(ws + off); off += (size_t)NA * QKV_STRIDE * 2;
    __bf16* vaT = (__bf16*)(ws + off); off += (size_t)NA * QKV_STRIDE * 2;
    float*  vaf = (float* )(ws + off); off += (size_t)NA * QKV_STRIDE * 4;
    __bf16* qb  = (__bf16*)(ws + off); off += (size_t)NB * QKV_STRIDE * 2;
    __bf16* kb  = (__bf16*)(ws + off); off += (size_t)NB * QKV_STRIDE * 2;
    __bf16* vbT = (__bf16*)(ws + off); off += (size_t)NB * QKV_STRIDE * 2;
    float*  vbf = (float* )(ws + off); off += (size_t)NB * QKV_STRIDE * 4;

    qkv_kernel<<<NA + NB, 256, 0, stream>>>(fa, fb, W,
                                            qa, ka, vaT, vaf, qb, kb, vbT, vbf);
    attn_kernel<<<dim3(NA, NB / 32), 128, 0, stream>>>(qa, ka, vaT, vaf,
                                                       qb, kb, vbT, vbf,
                                                       (float*)d_out);
}
